// QMIXNet_25357486916099
// MI455X (gfx1250) — matrix operand order repaired and checked by reading, not compiled, not run on hardware
//
#include <hip/hip_runtime.h>

// ---------------------------------------------------------------------------
// QMIX mixing network, fused, for MI455X (gfx1250, wave32, WMMA + async LDS).
//
// Roofline: 41 GFLOP / 66 MB => ~620 FLOP/byte. Memory floor at 23.3 TB/s is
// ~2.8us; f16 WMMA (16x16x32, f32 accum) keeps compute near that floor, so all
// six batched GEMMs go through v_wmma_f32_16x16x32_f16. Weights (~0.6 MB) are
// pre-packed into the ISA B-fragment layout in d_ws (L2-resident, coalesced
// 32B/lane loads). The dominant HBM stream (state, 64 MB) is staged into LDS
// with GLOBAL_LOAD_ASYNC_TO_LDS_B128 (ASYNCcnt path, no VGPR round trip).
// ---------------------------------------------------------------------------

typedef _Float16 v16h __attribute__((ext_vector_type(16)));
typedef _Float16 v8h  __attribute__((ext_vector_type(8)));
typedef float    v8f  __attribute__((ext_vector_type(8)));

#define STATE 256
#define KT    8      // 256 / 32 k-chunks per GEMM
#define WAVES 2      // waves per block; LDS: 2*(16KB f32 stage + 8KB f16 S) = 48KB

// Workspace offsets (in halves). Each fragment tile = 512 halves (32 lanes x 16).
#define O_W1A  0        // 256x256 : 8*16  tiles
#define O_W1B  65536    // 256x512 : 8*32  tiles
#define O_W2A  196608   // 256x256 : 8*16  tiles
#define O_W2B  262144   // 256x64  : 8*4   tiles
#define O_WB1  278528   // 256x64  : 8*4   tiles
#define O_WB2A 294912   // 256x64  : 8*4   tiles
// total 311296 halves = 622592 bytes of d_ws

// Pack one f32 weight matrix [K x Ncol] (row-major, K = fan_in) into f16
// B-matrix fragments: tile (kt, nt) covers K in [kt*32, +32), N in [nt*16, +16).
// ISA wave32 B layout (16-bit): lane l holds column n = l&15, K = (l>>4)*8 + (i&7) + (i>>3)*16.
__global__ void qmix_pack_w(const float* __restrict__ src, _Float16* __restrict__ dst,
                            int Ncol) {
  int tile   = blockIdx.x;
  int ntiles = Ncol >> 4;
  int kt = tile / ntiles;
  int nt = tile - kt * ntiles;
  int lane = threadIdx.x;
  int n     = nt * 16 + (lane & 15);
  int kbase = kt * 32 + ((lane >> 4) << 3);
  alignas(32) _Float16 tmp[16];
#pragma unroll
  for (int i = 0; i < 16; ++i) tmp[i] = (_Float16)src[(size_t)(kbase + (i & 7) + ((i >> 3) << 4)) * Ncol + n];
  *((v16h*)(dst + ((size_t)tile << 9) + (lane << 4))) = *(const v16h*)tmp;
}

__global__ __launch_bounds__(WAVES * 32)
void qmix_fused(const float* __restrict__ qv,   // [N][8]
                const float* __restrict__ sb,   // [N][256]
                const float* __restrict__ b1a,  // [256]
                const float* __restrict__ b1b,  // [512]
                const float* __restrict__ b2a,  // [256]
                const float* __restrict__ b2b,  // [64]
                const float* __restrict__ bb1,  // [64]
                const float* __restrict__ bb2a, // [64]
                const float* __restrict__ wb2b, // [64]
                const float* __restrict__ bb2b, // [1]
                const _Float16* __restrict__ wp,
                float* __restrict__ out) {
  // Per-wave: Sg = 16KB f32 async-staging (its first 8KB is re-used as the f16
  // hidden buffer once the f32 state has been converted); Sh = 8KB f16 state.
  __shared__ float    Sg[WAVES][16 * STATE];
  __shared__ _Float16 Sh[WAVES][16 * STATE];

  const int lane  = threadIdx.x & 31;
  const int w     = threadIdx.x >> 5;
  const int tile  = blockIdx.x * WAVES + w;   // grid sized exactly: no guard
  const int rowbase = tile * 16;
  const int nlane   = lane & 15;
  const int hi8     = (lane >> 4) << 3;       // row/K-half select per ISA layouts

  // Prefetch next tile-group's state rows into the cache hierarchy.
  __builtin_prefetch(sb + (size_t)(rowbase + WAVES * 16) * STATE, 0, 1);

  // ---- Async-stage S tile: 16x256 f32 -> LDS via GLOBAL_LOAD_ASYNC_TO_LDS ----
  {
    unsigned ldsBase = (unsigned)(uintptr_t)&Sg[w][0];   // low 32 bits = LDS offset
    const char* gBase = (const char*)(sb + (size_t)rowbase * STATE);
#pragma unroll 1
    for (int t = 0; t < 32; ++t) {           // 32 lanes * 16B * 32 iters = 16KB
      unsigned    loff = ldsBase + (unsigned)(t * 512 + lane * 16);
      const char* ga   = gBase + t * 512 + lane * 16;
      asm volatile("global_load_async_to_lds_b128 %0, %1, off"
                   :: "v"(loff), "v"(ga) : "memory");
    }
    asm volatile("s_wait_asynccnt 0x0" ::: "memory");
  }

  // ---- Convert f32 state (LDS) -> f16 state (LDS), 128 elems/lane ----
  {
    const float4* sg4 = (const float4*)&Sg[w][0];
    for (int t = lane; t < (16 * STATE / 4); t += 32) {
      float4 v = sg4[t];
      _Float16* d = &Sh[w][t * 4];
      d[0] = (_Float16)v.x; d[1] = (_Float16)v.y;
      d[2] = (_Float16)v.z; d[3] = (_Float16)v.w;
    }
  }
  __syncthreads();   // fences the aliased reuse of Sg as the f16 hidden buffer

  const _Float16* Sb = &Sh[w][0];
  _Float16*       Hb = (_Float16*)&Sg[w][0];   // 8KB of the dead f32 staging

  // A fragment (16x32 f16): lane holds row m=lane&15; halves [k0+hi8 .. +7] and
  // [k0+16+hi8 .. +7]  => two ds_load_b128.
  auto loadA = [&](const _Float16* base, int k0) {
    union { v16h v; v8h h[2]; } a;
    const _Float16* p = base + (nlane * STATE + k0 + hi8);
    a.h[0] = *(const v8h*)p;
    a.h[1] = *(const v8h*)(p + 16);
    return a.v;
  };
  // B fragment: pre-packed, 32 bytes per lane, contiguous.
  auto loadB = [&](const _Float16* wbase, int ntiles, int kt, int nt) {
    return *((const v16h*)wbase + ((kt * ntiles + nt) * 32 + lane));
  };

  // ---- GEMM1: H1 = relu(S @ W1a + b1a)  -> Hb (f16) ----
#pragma unroll 1
  for (int nt = 0; nt < 16; ++nt) {
    float bv = b1a[nt * 16 + nlane];
    v8f acc; for (int i = 0; i < 8; ++i) acc[i] = bv;
    for (int kt = 0; kt < KT; ++kt) {
      v16h a = loadA(Sb, kt * 32);
      v16h b = loadB(wp + O_W1A, 16, kt, nt);
      acc = __builtin_amdgcn_wmma_f32_16x16x32_f16(false, a, false, b, (short)0, acc, false, false);
    }
    for (int i = 0; i < 8; ++i) {
      float v = acc[i]; v = v > 0.f ? v : 0.f;
      Hb[(hi8 + i) * STATE + nt * 16 + nlane] = (_Float16)v;
    }
  }

  // ---- b1 tiles: hid = S @ Wb1 + bb1  (D-layout, q-cols in 4 tiles of 16) ----
  v8f hid[4];
#pragma unroll
  for (int t = 0; t < 4; ++t) {
    float bv = bb1[t * 16 + nlane];
    v8f acc; for (int i = 0; i < 8; ++i) acc[i] = bv;
    for (int kt = 0; kt < KT; ++kt) {
      v16h a = loadA(Sb, kt * 32);
      v16h b = loadB(wp + O_WB1, 4, kt, t);
      acc = __builtin_amdgcn_wmma_f32_16x16x32_f16(false, a, false, b, (short)0, acc, false, false);
    }
    hid[t] = acc;
  }

  // ---- GEMM2: w1 = |H1 @ W1b + b1b| ; hid += q[.,agent] * w1 tile ----
#pragma unroll 1
  for (int ag = 0; ag < 8; ++ag) {
    float qrow[8];
#pragma unroll
    for (int i = 0; i < 8; ++i) qrow[i] = qv[(size_t)(rowbase + hi8 + i) * 8 + ag];
#pragma unroll
    for (int qt = 0; qt < 4; ++qt) {
      int j = ag * 4 + qt;                 // output col tile of the 512-wide w1
      float bv = b1b[j * 16 + nlane];
      v8f acc; for (int i = 0; i < 8; ++i) acc[i] = bv;
      for (int kt = 0; kt < KT; ++kt) {
        v16h a = loadA(Hb, kt * 32);
        v16h b = loadB(wp + O_W1B, 32, kt, j);
        acc = __builtin_amdgcn_wmma_f32_16x16x32_f16(false, a, false, b, (short)0, acc, false, false);
      }
      for (int i = 0; i < 8; ++i) hid[qt][i] += qrow[i] * __builtin_fabsf(acc[i]);
    }
  }
  // elu
#pragma unroll
  for (int t = 0; t < 4; ++t)
    for (int i = 0; i < 8; ++i) {
      float h = hid[t][i];
      hid[t][i] = h > 0.f ? h : (__expf(h) - 1.f);
    }

  // ---- H2 = relu(S @ W2a + b2a) -> Hb (reuse buffer) ----
#pragma unroll 1
  for (int nt = 0; nt < 16; ++nt) {
    float bv = b2a[nt * 16 + nlane];
    v8f acc; for (int i = 0; i < 8; ++i) acc[i] = bv;
    for (int kt = 0; kt < KT; ++kt) {
      v16h a = loadA(Sb, kt * 32);
      v16h b = loadB(wp + O_W2A, 16, kt, nt);
      acc = __builtin_amdgcn_wmma_f32_16x16x32_f16(false, a, false, b, (short)0, acc, false, false);
    }
    for (int i = 0; i < 8; ++i) {
      float v = acc[i]; v = v > 0.f ? v : 0.f;
      Hb[(hi8 + i) * STATE + nt * 16 + nlane] = (_Float16)v;
    }
  }

  // ---- w2 = |H2 @ W2b + b2b| ----
  v8f w2t[4];
#pragma unroll
  for (int t = 0; t < 4; ++t) {
    float bv = b2b[t * 16 + nlane];
    v8f acc; for (int i = 0; i < 8; ++i) acc[i] = bv;
    for (int kt = 0; kt < KT; ++kt) {
      v16h a = loadA(Hb, kt * 32);
      v16h b = loadB(wp + O_W2B, 4, kt, t);
      acc = __builtin_amdgcn_wmma_f32_16x16x32_f16(false, a, false, b, (short)0, acc, false, false);
    }
    for (int i = 0; i < 8; ++i) w2t[t][i] = __builtin_fabsf(acc[i]);
  }

  // ---- b2 path: relu(S @ Wb2a + bb2a) dotted with Wb2b ; plus hid . w2 ----
  float r[8];
#pragma unroll
  for (int i = 0; i < 8; ++i) r[i] = 0.f;
#pragma unroll
  for (int t = 0; t < 4; ++t) {
    float bv = bb2a[t * 16 + nlane];
    v8f acc; for (int i = 0; i < 8; ++i) acc[i] = bv;
    for (int kt = 0; kt < KT; ++kt) {
      v16h a = loadA(Sb, kt * 32);
      v16h b = loadB(wp + O_WB2A, 4, kt, t);
      acc = __builtin_amdgcn_wmma_f32_16x16x32_f16(false, a, false, b, (short)0, acc, false, false);
    }
    float ws = wb2b[t * 16 + nlane];
    for (int i = 0; i < 8; ++i) {
      float h = acc[i]; h = h > 0.f ? h : 0.f;
      r[i] += h * ws;
    }
    for (int i = 0; i < 8; ++i) r[i] += hid[t][i] * w2t[t][i];
  }

  // Reduce over the 16 N-lanes of each half-wave (rows live per-half).
#pragma unroll
  for (int i = 0; i < 8; ++i) {
    r[i] += __shfl_xor(r[i], 1, 32);
    r[i] += __shfl_xor(r[i], 2, 32);
    r[i] += __shfl_xor(r[i], 4, 32);
    r[i] += __shfl_xor(r[i], 8, 32);
  }
  if (nlane == 0) {
    float bb = bb2b[0];
#pragma unroll
    for (int i = 0; i < 8; ++i) out[rowbase + hi8 + i] = r[i] + bb;
  }
}

extern "C" void kernel_launch(void* const* d_in, const int* in_sizes, int n_in,
                              void* d_out, int out_size, void* d_ws, size_t ws_size,
                              hipStream_t stream) {
  const float* qv   = (const float*)d_in[0];
  const float* sb   = (const float*)d_in[1];
  const float* W1a  = (const float*)d_in[2];
  const float* b1a  = (const float*)d_in[3];
  const float* W1b  = (const float*)d_in[4];
  const float* b1b  = (const float*)d_in[5];
  const float* W2a  = (const float*)d_in[6];
  const float* b2a  = (const float*)d_in[7];
  const float* W2b  = (const float*)d_in[8];
  const float* b2b  = (const float*)d_in[9];
  const float* Wb1  = (const float*)d_in[10];
  const float* bb1  = (const float*)d_in[11];
  const float* Wb2a = (const float*)d_in[12];
  const float* bb2a = (const float*)d_in[13];
  const float* Wb2b = (const float*)d_in[14];
  const float* bb2b = (const float*)d_in[15];

  _Float16* wp = (_Float16*)d_ws;   // 622592 bytes used

  // Pack all six weight matrices into f16 WMMA B-fragment layout (one wave/tile).
  qmix_pack_w<<<8 * 16, 32, 0, stream>>>(W1a,  wp + O_W1A,  256);
  qmix_pack_w<<<8 * 32, 32, 0, stream>>>(W1b,  wp + O_W1B,  512);
  qmix_pack_w<<<8 * 16, 32, 0, stream>>>(W2a,  wp + O_W2A,  256);
  qmix_pack_w<<<8 * 4,  32, 0, stream>>>(W2b,  wp + O_W2B,  64);
  qmix_pack_w<<<8 * 4,  32, 0, stream>>>(Wb1,  wp + O_WB1,  64);
  qmix_pack_w<<<8 * 4,  32, 0, stream>>>(Wb2a, wp + O_WB2A, 64);

  int N = in_sizes[1] / STATE;          // 65536 samples
  int ntiles = N / 16;                  // 4096 row tiles
  int blocks = ntiles / WAVES;          // exact: 4096 % WAVES == 0
  qmix_fused<<<blocks, WAVES * 32, 0, stream>>>(
      qv, sb, b1a, b1b, b2a, b2b, bb1, bb2a, Wb2b, bb2b,
      wp, (float*)d_out);
}
